// GalaxyLLM_34548716929745
// MI455X (gfx1250) — matrix operand order repaired and checked by reading, not compiled, not run on hardware
//
#include <hip/hip_runtime.h>
#include <math.h>

typedef __attribute__((ext_vector_type(16))) _Float16 v16h;
typedef __attribute__((ext_vector_type(8)))  float    v8f;

// Problem constants (from reference setup_inputs)
#define KB   4        // batch
#define KS   512      // sequence
#define KDM  1024     // d_model
#define KL   12       // layers
#define KV   32000    // vocab
#define KE   8        // experts
#define KF   4096     // ffn dim
#define KH   16       // heads
#define KDH  64       // head dim
#define KT   (KB*KS)  // tokens = 2048
#define KTD  ((long long)KT*KDM)

// ---------------------------------------------------------------------------
// Operand fragment loaders: fp32 global -> f16 WMMA operand registers.
// A operand per-lane layout (16-bit A 16x32): lane l holds row m=l&15;
//   halves 0..15 -> K = k0 + 2*hi8 + {0..15},     hi8 = (l>=16)?8:0
// B operand per-lane layout (16-bit B 32x16): lane l holds col n=l&15;
//   halves 0..15 -> K = k0 + hi16 + {0..15},      hi16 = (l>=16)?16:0
// ---------------------------------------------------------------------------
__device__ __forceinline__ v16h load_cvt_a(const float* __restrict__ row, int k0, int hi8)
{
    const float4 a0 = *(const float4*)(row + k0 + 2 * hi8);
    const float4 a1 = *(const float4*)(row + k0 + 2 * hi8 + 4);
    const float4 a2 = *(const float4*)(row + k0 + 2 * hi8 + 8);
    const float4 a3 = *(const float4*)(row + k0 + 2 * hi8 + 12);
    v16h v;
    v[0]=(_Float16)a0.x;  v[1]=(_Float16)a0.y;  v[2]=(_Float16)a0.z;  v[3]=(_Float16)a0.w;
    v[4]=(_Float16)a1.x;  v[5]=(_Float16)a1.y;  v[6]=(_Float16)a1.z;  v[7]=(_Float16)a1.w;
    v[8]=(_Float16)a2.x;  v[9]=(_Float16)a2.y;  v[10]=(_Float16)a2.z; v[11]=(_Float16)a2.w;
    v[12]=(_Float16)a3.x; v[13]=(_Float16)a3.y; v[14]=(_Float16)a3.z; v[15]=(_Float16)a3.w;
    return v;
}

__device__ __forceinline__ v16h load_cvt_b(const float* __restrict__ row, int k0, int hi16)
{
    const float4 b0 = *(const float4*)(row + k0 + hi16);
    const float4 b1 = *(const float4*)(row + k0 + hi16 + 4);
    const float4 b2 = *(const float4*)(row + k0 + hi16 + 8);
    const float4 b3 = *(const float4*)(row + k0 + hi16 + 12);
    v16h v;
    v[0]=(_Float16)b0.x;  v[1]=(_Float16)b0.y;  v[2]=(_Float16)b0.z;  v[3]=(_Float16)b0.w;
    v[4]=(_Float16)b1.x;  v[5]=(_Float16)b1.y;  v[6]=(_Float16)b1.z;  v[7]=(_Float16)b1.w;
    v[8]=(_Float16)b2.x;  v[9]=(_Float16)b2.y;  v[10]=(_Float16)b2.z; v[11]=(_Float16)b2.w;
    v[12]=(_Float16)b3.x; v[13]=(_Float16)b3.y; v[14]=(_Float16)b3.z; v[15]=(_Float16)b3.w;
    return v;
}

#define WMMA_F16(A_, B_, C_) \
    __builtin_amdgcn_wmma_f32_16x16x32_f16(false, (A_), false, (B_), (short)0, (C_), false, false)

// ---------------------------------------------------------------------------
// Register-tiled WMMA GEMM: C[M,N] = act(alpha * A[M,K] x W[N,K]^T + bias[N])
// Requires M % 32 == 0, N % 64 == 0, K % 32 == 0 (true for every GEMM here).
// Block = 8 waves in 2(M) x 4(N); each wave computes a 32x64 patch as
// 2x4 WMMA 16x16 tiles: per K-step 24 loads feed 8 v_wmma (3 loads/wmma),
// with 8 independent accumulator chains to hide XDL latency.
// Batched via blockIdx.z with element strides sA/sW/sC.
// ---------------------------------------------------------------------------
__global__ __launch_bounds__(256) void k_gemm_f16(
    const float* __restrict__ A, const float* __restrict__ W,
    const float* __restrict__ bias, float* __restrict__ C,
    int M, int N, int K,
    long long sA, long long sW, long long sC,
    float alpha, int act)
{
    const int lane  = threadIdx.x & 31;
    const int wave  = threadIdx.x >> 5;
    const int waveM = wave >> 2;   // 0..1
    const int waveN = wave & 3;    // 0..3

    const int m0 = (blockIdx.y * 2 + waveM) * 32;   // wave's first output row
    const int n0 = (blockIdx.x * 4 + waveN) * 64;   // wave's first output col
    if (m0 >= M || n0 >= N) return;                 // whole-wave uniform exit

    const long long bz = blockIdx.z;
    A += bz * sA;
    W += bz * sW;
    C += bz * sC;

    const int l16  = lane & 15;
    const int hi   = lane >> 4;   // 0 or 1
    const int hi8  = hi * 8;
    const int hi16 = hi * 16;

    const float* __restrict__ arow0 = A + (long long)(m0 + l16) * K;
    const float* __restrict__ arow1 = A + (long long)(m0 + 16 + l16) * K;
    const float* __restrict__ wrow0 = W + (long long)(n0 + l16) * K;
    const float* __restrict__ wrow1 = W + (long long)(n0 + 16 + l16) * K;
    const float* __restrict__ wrow2 = W + (long long)(n0 + 32 + l16) * K;
    const float* __restrict__ wrow3 = W + (long long)(n0 + 48 + l16) * K;

    v8f acc[2][4];
    #pragma unroll
    for (int i = 0; i < 2; ++i)
        #pragma unroll
        for (int j = 0; j < 4; ++j)
            acc[i][j] = (v8f){0.f, 0.f, 0.f, 0.f, 0.f, 0.f, 0.f, 0.f};

    for (int k0 = 0; k0 < K; k0 += 32) {
        // pull the K-block two steps ahead of A toward the WGP while WMMAs run
        __builtin_prefetch(arow0 + k0 + 64, 0, 1);
        __builtin_prefetch(arow1 + k0 + 64, 0, 1);

        const v16h a0 = load_cvt_a(arow0, k0, hi8);
        const v16h a1 = load_cvt_a(arow1, k0, hi8);
        const v16h b0 = load_cvt_b(wrow0, k0, hi16);
        const v16h b1 = load_cvt_b(wrow1, k0, hi16);
        const v16h b2 = load_cvt_b(wrow2, k0, hi16);
        const v16h b3 = load_cvt_b(wrow3, k0, hi16);

        acc[0][0] = WMMA_F16(a0, b0, acc[0][0]);
        acc[0][1] = WMMA_F16(a0, b1, acc[0][1]);
        acc[0][2] = WMMA_F16(a0, b2, acc[0][2]);
        acc[0][3] = WMMA_F16(a0, b3, acc[0][3]);
        acc[1][0] = WMMA_F16(a1, b0, acc[1][0]);
        acc[1][1] = WMMA_F16(a1, b1, acc[1][1]);
        acc[1][2] = WMMA_F16(a1, b2, acc[1][2]);
        acc[1][3] = WMMA_F16(a1, b3, acc[1][3]);
    }

    // Epilogue. C/D layout: VGPR r holds (row = tile_m + hi8 + r, col = tile_n + l16)
    #pragma unroll
    for (int j = 0; j < 4; ++j) {
        const int   n  = n0 + j * 16 + l16;
        const float bb = bias ? bias[n] : 0.0f;
        #pragma unroll
        for (int i = 0; i < 2; ++i) {
            const int row0 = m0 + i * 16 + hi8;
            #pragma unroll
            for (int r = 0; r < 8; ++r) {
                float v = acc[i][j][r] * alpha + bb;
                if (act == 1) v = 0.5f * v * (1.0f + erff(v * 0.70710678118654752f)); // exact GELU
                C[(long long)(row0 + r) * N + n] = v;
            }
        }
    }
}

// ---------------------------------------------------------------------------
// Embedding: x[t,d] = tok_emb[ids[t],d] + pos_emb[t%S,d]
// ---------------------------------------------------------------------------
__global__ __launch_bounds__(256) void k_embed(
    const int* __restrict__ ids, const float* __restrict__ tok,
    const float* __restrict__ pos, float* __restrict__ x)
{
    long long i = (long long)blockIdx.x * 256 + threadIdx.x;
    if (i >= KTD) return;
    int t = (int)(i / KDM), d = (int)(i % KDM);
    int s = t % KS;
    x[i] = tok[(long long)ids[t] * KDM + d] + pos[(long long)s * KDM + d];
}

// ---------------------------------------------------------------------------
// Split qkv[T,3D] into Q[b,h,s,dh], K[b,h,s,dh], V^T[b,h,dh,s]
// ---------------------------------------------------------------------------
__global__ __launch_bounds__(256) void k_split_heads(
    const float* __restrict__ qkv, float* __restrict__ Q,
    float* __restrict__ Kh, float* __restrict__ Vt)
{
    long long i = (long long)blockIdx.x * 256 + threadIdx.x;
    if (i >= KTD) return;
    int t = (int)(i / KDM), d = (int)(i % KDM);
    int h = d / KDH, di = d % KDH;
    int b = t / KS,  s  = t % KS;
    long long base = (long long)t * (3 * KDM);
    long long bh = (long long)b * KH + h;
    Q [(bh * KS + s) * KDH + di] = qkv[base + d];
    Kh[(bh * KS + s) * KDH + di] = qkv[base + KDM + d];
    Vt[(bh * KDH + di) * KS + s] = qkv[base + 2 * KDM + d];
}

// ctx[b,h,s,dh] -> y[t, h*dh+di]
__global__ __launch_bounds__(256) void k_merge_heads(
    const float* __restrict__ ctx, float* __restrict__ y)
{
    long long i = (long long)blockIdx.x * 256 + threadIdx.x;
    if (i >= KTD) return;
    int t = (int)(i / KDM), d = (int)(i % KDM);
    int h = d / KDH, di = d % KDH;
    int b = t / KS,  s  = t % KS;
    long long bh = (long long)b * KH + h;
    y[i] = ctx[(bh * KS + s) * KDH + di];
}

// ---------------------------------------------------------------------------
// In-place row softmax (row length = rowlen), one 256-thread block per row
// ---------------------------------------------------------------------------
__global__ __launch_bounds__(256) void k_softmax(float* __restrict__ p, int rowlen)
{
    float* row = p + (long long)blockIdx.x * rowlen;
    __shared__ float red[256];
    const int tid = threadIdx.x;

    float m = -3.4e38f;
    for (int i = tid; i < rowlen; i += 256) m = fmaxf(m, row[i]);
    red[tid] = m; __syncthreads();
    for (int s = 128; s > 0; s >>= 1) {
        if (tid < s) red[tid] = fmaxf(red[tid], red[tid + s]);
        __syncthreads();
    }
    const float mx = red[0];
    __syncthreads();

    float sum = 0.f;
    for (int i = tid; i < rowlen; i += 256) sum += expf(row[i] - mx);
    red[tid] = sum; __syncthreads();
    for (int s = 128; s > 0; s >>= 1) {
        if (tid < s) red[tid] += red[tid + s];
        __syncthreads();
    }
    const float inv = 1.0f / red[0];

    for (int i = tid; i < rowlen; i += 256) row[i] = expf(row[i] - mx) * inv;
}

// ---------------------------------------------------------------------------
// y = LayerNorm(x + r) * g + b   (row length KDM=1024, block=256 -> 4/thread)
// ---------------------------------------------------------------------------
__global__ __launch_bounds__(256) void k_layernorm(
    const float* __restrict__ x, const float* __restrict__ r,
    const float* __restrict__ g, const float* __restrict__ b,
    float* __restrict__ y)
{
    const long long t = blockIdx.x;
    const float* xr = x + t * KDM;
    const float* rr = r + t * KDM;
    __shared__ float s1[256], s2[256];
    const int tid = threadIdx.x;

    float vals[4];
    float a = 0.f, q = 0.f;
    #pragma unroll
    for (int j = 0; j < 4; ++j) {
        int d = tid + j * 256;
        float v = xr[d] + rr[d];
        vals[j] = v;
        a += v; q += v * v;
    }
    s1[tid] = a; s2[tid] = q; __syncthreads();
    for (int s = 128; s > 0; s >>= 1) {
        if (tid < s) { s1[tid] += s1[tid + s]; s2[tid] += s2[tid + s]; }
        __syncthreads();
    }
    const float mean = s1[0] * (1.0f / KDM);
    const float var  = s2[0] * (1.0f / KDM) - mean * mean;
    const float rstd = rsqrtf(var + 1e-5f);

    #pragma unroll
    for (int j = 0; j < 4; ++j) {
        int d = tid + j * 256;
        y[t * KDM + d] = (vals[j] - mean) * rstd * g[d] + b[d];
    }
}

// ---------------------------------------------------------------------------
// Router: one wave per token. logits[e] = x[t]·Wr[e] + br[e]; softmax over 8;
// top-2 (k=2 in reference) renormalized -> dense combine weights comb[T,E].
// ---------------------------------------------------------------------------
__global__ __launch_bounds__(256) void k_router(
    const float* __restrict__ x, const float* __restrict__ Wr,
    const float* __restrict__ br, float* __restrict__ comb)
{
    const int lane = threadIdx.x & 31;
    const int wave = threadIdx.x >> 5;
    const int t = blockIdx.x * 8 + wave;
    if (t >= KT) return;
    const float* h = x + (long long)t * KDM;

    float logit[KE];
    #pragma unroll
    for (int e = 0; e < KE; ++e) {
        float p = 0.f;
        for (int d = lane; d < KDM; d += 32) p += h[d] * Wr[(long long)e * KDM + d];
        #pragma unroll
        for (int off = 16; off > 0; off >>= 1) p += __shfl_xor(p, off, 32);
        logit[e] = p + br[e];
    }
    if (lane == 0) {
        float mx = logit[0];
        for (int e = 1; e < KE; ++e) mx = fmaxf(mx, logit[e]);
        float p[KE], s = 0.f;
        for (int e = 0; e < KE; ++e) { p[e] = expf(logit[e] - mx); s += p[e]; }
        for (int e = 0; e < KE; ++e) p[e] /= s;
        int i1 = 0;
        for (int e = 1; e < KE; ++e) if (p[e] > p[i1]) i1 = e;
        int i2 = (i1 == 0) ? 1 : 0;
        for (int e = 0; e < KE; ++e) if (e != i1 && p[e] > p[i2]) i2 = e;
        const float wsum = p[i1] + p[i2];
        for (int e = 0; e < KE; ++e)
            comb[(long long)t * KE + e] =
                (e == i1) ? p[i1] / wsum : ((e == i2) ? p[i2] / wsum : 0.f);
    }
}

// moe[t,d] = (first ? 0 : moe[t,d]) + comb[t,e] * eout[t,d]
__global__ __launch_bounds__(256) void k_moe_accum(
    const float* __restrict__ eout, const float* __restrict__ comb,
    float* __restrict__ moe, int e, int first)
{
    long long i = (long long)blockIdx.x * 256 + threadIdx.x;
    if (i >= KTD) return;
    int t = (int)(i / KDM);
    float w = comb[(long long)t * KE + e];
    float base = first ? 0.f : moe[i];
    moe[i] = base + w * eout[i];
}

// ---------------------------------------------------------------------------
// Host side
// ---------------------------------------------------------------------------
static inline void gemm(hipStream_t st,
                        const float* A, const float* W, const float* bias, float* C,
                        int M, int N, int K, int batch,
                        long long sA, long long sW, long long sC,
                        float alpha, int act)
{
    dim3 g((N + 255) / 256, (M + 63) / 64, batch);
    k_gemm_f16<<<g, 256, 0, st>>>(A, W, bias, C, M, N, K, sA, sW, sC, alpha, act);
}

extern "C" void kernel_launch(void* const* d_in, const int* in_sizes, int n_in,
                              void* d_out, int out_size, void* d_ws, size_t ws_size,
                              hipStream_t stream)
{
    (void)in_sizes; (void)n_in; (void)out_size; (void)ws_size;

    const int*   ids   = (const int*)d_in[0];
    // d_in[1] = k (top-k count, == 2; hardcoded in k_router)
    const float* tok   = (const float*)d_in[2];
    const float* pos   = (const float*)d_in[3];
    const float* Wqkv  = (const float*)d_in[4];
    const float* bqkv  = (const float*)d_in[5];
    const float* Wo    = (const float*)d_in[6];
    const float* bo    = (const float*)d_in[7];
    const float* ln1g  = (const float*)d_in[8];
    const float* ln1b  = (const float*)d_in[9];
    const float* ln2g  = (const float*)d_in[10];
    const float* ln2b  = (const float*)d_in[11];
    const float* W1    = (const float*)d_in[12];
    const float* b1    = (const float*)d_in[13];
    const float* W2    = (const float*)d_in[14];
    const float* b2    = (const float*)d_in[15];
    const float* Wr    = (const float*)d_in[16];
    const float* br    = (const float*)d_in[17];
    const float* We1   = (const float*)d_in[18];
    const float* be1   = (const float*)d_in[19];
    const float* We2   = (const float*)d_in[20];
    const float* be2   = (const float*)d_in[21];
    const float* Wout  = (const float*)d_in[22];
    const float* bout  = (const float*)d_in[23];

    float* ws = (float*)d_ws;
    // Workspace layout (units of KTD = 2M floats unless noted)
    float* x      = ws;                 // [T,D]
    float* y      = ws + 1 * KTD;       // [T,D]
    float* qkv    = ws + 2 * KTD;       // [T,3D]  (3 units)
    float* Q      = ws + 5 * KTD;       // [BH,S,dh]
    float* Kh     = ws + 6 * KTD;       // [BH,S,dh]
    float* Vt     = ws + 7 * KTD;       // [BH,dh,S]
    float* ctx    = ws + 8 * KTD;       // [BH,S,dh]
    float* ctxm   = ws + 9 * KTD;       // [T,D]
    float* attn   = ws + 10 * KTD;      // [T,D]   (attn-out / ffn-out scratch)
    float* ffh    = ws + 11 * KTD;      // [T,F]   (4 units)
    float* scores = ws + 15 * KTD;      // [BH,S,S] (8 units)
    float* moe    = ws + 23 * KTD;      // [T,D]
    float* eout   = ws + 24 * KTD;      // [T,D]
    float* comb   = ws + 25 * KTD;      // [T,E]

    const int gTD = (int)((KTD + 255) / 256);
    const float isqrt_dh = 0.125f; // 1/sqrt(64)

    // Embedding
    k_embed<<<gTD, 256, 0, stream>>>(ids, tok, pos, x);

    // Encoder layers (post-norm, exact-GELU FFN)
    for (int l = 0; l < KL; ++l) {
        const long long o3DD = (long long)l * 3 * KDM * KDM;
        const long long oDD  = (long long)l * KDM * KDM;
        const long long oFD  = (long long)l * KF * KDM;
        const long long oD   = (long long)l * KDM;
        const long long o3D  = (long long)l * 3 * KDM;
        const long long oF   = (long long)l * KF;

        // qkv = x @ Wqkv^T + bqkv
        gemm(stream, x, Wqkv + o3DD, bqkv + o3D, qkv, KT, 3 * KDM, KDM, 1, 0, 0, 0, 1.0f, 0);
        k_split_heads<<<gTD, 256, 0, stream>>>(qkv, Q, Kh, Vt);

        // scores = (Q K^T) / sqrt(dh), batched over B*H
        gemm(stream, Q, Kh, nullptr, scores, KS, KS, KDH, KB * KH,
             (long long)KS * KDH, (long long)KS * KDH, (long long)KS * KS, isqrt_dh, 0);
        k_softmax<<<KB * KH * KS, 256, 0, stream>>>(scores, KS);

        // ctx = attn @ V   (V stored transposed -> fits A·W^T form)
        gemm(stream, scores, Vt, nullptr, ctx, KS, KDH, KS, KB * KH,
             (long long)KS * KS, (long long)KDH * KS, (long long)KS * KDH, 1.0f, 0);
        k_merge_heads<<<gTD, 256, 0, stream>>>(ctx, ctxm);

        // attn_out = ctx @ Wo^T + bo ; x1 = LN(x + attn_out)
        gemm(stream, ctxm, Wo + oDD, bo + oD, attn, KT, KDM, KDM, 1, 0, 0, 0, 1.0f, 0);
        k_layernorm<<<KT, 256, 0, stream>>>(x, attn, ln1g + oD, ln1b + oD, y);

        // ff = gelu(y @ W1^T + b1) @ W2^T + b2 ; x = LN(y + ff)
        gemm(stream, y, W1 + oFD, b1 + oF, ffh, KT, KF, KDM, 1, 0, 0, 0, 1.0f, 1);
        gemm(stream, ffh, W2 + oFD, b2 + oD, attn, KT, KDM, KF, 1, 0, 0, 0, 1.0f, 0);
        k_layernorm<<<KT, 256, 0, stream>>>(y, attn, ln2g + oD, ln2b + oD, x);
    }

    // Router: softmax(x@Wr^T+br), top-2 renormalized -> comb[T,E]
    k_router<<<KT / 8, 256, 0, stream>>>(x, Wr, br, comb);

    // Experts (dense, matching reference), combined with routing weights
    for (int e = 0; e < KE; ++e) {
        const long long oFD = (long long)e * KF * KDM;
        const long long oDF = (long long)e * KDM * KF;
        gemm(stream, x, We1 + oFD, be1 + (long long)e * KF, ffh, KT, KF, KDM, 1, 0, 0, 0, 1.0f, 1);
        gemm(stream, ffh, We2 + oDF, be2 + (long long)e * KDM, eout, KT, KDM, KF, 1, 0, 0, 0, 1.0f, 0);
        k_moe_accum<<<gTD, 256, 0, stream>>>(eout, comb, moe, e, e == 0 ? 1 : 0);
    }

    // logits = moe @ Wout^T + bout  -> d_out [B,S,V] fp32
    gemm(stream, moe, Wout, bout, (float*)d_out, KT, KV, KDM, 1, 0, 0, 0, 1.0f, 0);
}
